// Decoder_73229192397166
// MI455X (gfx1250) — compile-verified, matrix operand order checked by reading, not run on hardware
//
#include <hip/hip_runtime.h>
#include <hip/hip_bf16.h>

typedef __attribute__((ext_vector_type(16))) __bf16 v16bf;
typedef __attribute__((ext_vector_type(8)))  __bf16 v8bf;
typedef __attribute__((ext_vector_type(8)))  float  v8f;

#define B_  64
#define L_  128
#define H_  1024
#define T_  256
#define NBT 4       // batch tiles of 16 (WMMA M)
#define NHC 16      // hidden chunks of 64 units (N=256 gate cols per WG)
#define HPAD 1032   // padded h row in LDS: 1024 + 8 halfs (516 dwords, %64 = 4)

__device__ __forceinline__ unsigned short f2bf(float x) {
  unsigned int u = __float_as_uint(x);
  u += 0x7fffu + ((u >> 16) & 1u);      // round-to-nearest-even
  return (unsigned short)(u >> 16);
}

__device__ __forceinline__ float frcp(float x) {
#if __has_builtin(__builtin_amdgcn_rcpf)
  return __builtin_amdgcn_rcpf(x);
#else
  return 1.0f / x;
#endif
}
__device__ __forceinline__ float sigm(float x) { return frcp(1.0f + __expf(-x)); }
__device__ __forceinline__ float ftanh(float x) {
#if __has_builtin(__builtin_amdgcn_tanhf)
  return __builtin_amdgcn_tanhf(x);      // v_tanh_f32 (gfx1250 TRANS op)
#else
  float e = __expf(2.0f * x);
  return 1.0f - 2.0f * frcp(e + 1.0f);
#endif
}

// ---------------- init: zero sync counters + h double-buffer "read" side ----
__global__ void k_init(int* cnt, unsigned short* h0) {
  int i = blockIdx.x * blockDim.x + threadIdx.x;
  if (i < NBT) cnt[i] = 0;
  for (int j = i; j < B_ * H_; j += gridDim.x * blockDim.x) h0[j] = 0;
}

// ---------------- hidden = latent @ W_lin.T + b_lin  (64x1024, K=128) -------
__global__ void k_hidden(const float* __restrict__ latent, const float* __restrict__ W_lin,
                         const float* __restrict__ b_lin, float* __restrict__ hidden) {
  int i = blockIdx.x * 256 + threadIdx.x;          // 0..65535
  int b = i >> 10, h = i & 1023;
  const float4* l4 = (const float4*)(latent + b * L_);
  const float4* w4 = (const float4*)(W_lin + (size_t)h * L_);
  float s = b_lin[h];
  #pragma unroll 4
  for (int k = 0; k < L_ / 4; ++k) {
    float4 a = l4[k], w = w4[k];
    s += a.x * w.x + a.y * w.y + a.z * w.z + a.w * w.w;
  }
  hidden[i] = s;
}

// ---------------- x_gates = hidden @ W_ih.T + (b_ih + b_hh)  (64x4096) ------
__global__ void k_xgates(const float* __restrict__ hidden, const float* __restrict__ W_ih,
                         const float* __restrict__ b_ih, const float* __restrict__ b_hh,
                         float* __restrict__ xg) {
  int i = blockIdx.x * 256 + threadIdx.x;          // 0..262143
  int b = i >> 12, n = i & 4095;
  const float4* h4 = (const float4*)(hidden + (size_t)b * H_);
  const float4* w4 = (const float4*)(W_ih + (size_t)n * H_);
  float s = b_ih[n] + b_hh[n];
  #pragma unroll 4
  for (int k = 0; k < H_ / 4; ++k) {
    float4 a = h4[k], w = w4[k];
    s += a.x * w.x + a.y * w.y + a.z * w.z + a.w * w.w;
  }
  xg[i] = s;
}

// ---------------- pack W_hh (fp32 row-major) -> bf16 WMMA-B lane layout -----
// layout: [hc(16)][t(16)][kt(32)][lane(32)][e(16)]  (ushort each)
// tile t = g*4 + sub covers gate g, local units sub*16..+15 of chunk hc
// lane: N = lane&15 ; K-half select = lane>>4
// e:   K offset = (lane>>4)*8 + (e<8 ? e : e+8)   within kt*32
__global__ void k_pack(const float* __restrict__ W_hh, unsigned short* __restrict__ wpk) {
  size_t idx = (size_t)blockIdx.x * 256 + threadIdx.x;   // < 4194304
  int e    = (int)(idx & 15);
  int lane = (int)((idx >> 4) & 31);
  int kt   = (int)((idx >> 9) & 31);
  int t    = (int)((idx >> 14) & 15);
  int hc   = (int)(idx >> 18);
  int g = t >> 2, sub = t & 3, nn = lane & 15;
  int ggi  = g * H_ + hc * 64 + sub * 16 + nn;           // global gate row of W_hh
  int koff = ((lane >> 4) << 3) + (e < 8 ? e : e + 8);
  int k    = kt * 32 + koff;
  wpk[idx] = f2bf(W_hh[(size_t)ggi * H_ + k]);
}

// ---------------- persistent LSTM: 64 WGs, per-batch-tile barrier per step --
__global__ void __launch_bounds__(256, 1)
k_lstm(const float* __restrict__ xg, const unsigned short* __restrict__ wpk,
       const float* __restrict__ W_out, unsigned short* __restrict__ h0,
       unsigned short* __restrict__ h1, float* __restrict__ partial,
       int* __restrict__ cnt) {
  __shared__ __align__(16) unsigned short s_h[16 * HPAD];   // staged h tile (33 KB)
  __shared__ float s_gt[16][257];
  __shared__ float s_red[16][16];

  const int tid  = threadIdx.x;
  const int lane = tid & 31;
  const int wv   = tid >> 5;              // 0..7 (wave32)
  const int hc   = blockIdx.x;            // hidden chunk 0..15
  const int bt   = blockIdx.y;            // batch tile 0..3
  const int m0   = bt * 16;
  const int j0   = hc * 64;

  const int t0 = wv * 2, t1 = t0 + 1;
  const int nn = lane & 15;
  const int hi = lane >> 4;
  const unsigned short* wp0 = wpk + (size_t)(hc * 16 + t0) * 32 * 512 + lane * 16;
  const unsigned short* wp1 = wpk + (size_t)(hc * 16 + t1) * 32 * 512 + lane * 16;

  // Accumulator-init values (x_gates + biases) are loop constants: keep in regs.
  v8f xgr0{}, xgr1{};
  {
    int g0 = t0 >> 2, s0 = t0 & 3, g1 = t1 >> 2, s1 = t1 & 3;
    #pragma unroll
    for (int r = 0; r < 8; ++r) {
      int rr = r + hi * 8;
      xgr0[r] = xg[(size_t)(m0 + rr) * 4096 + (size_t)g0 * H_ + j0 + s0 * 16 + nn];
      xgr1[r] = xg[(size_t)(m0 + rr) * 4096 + (size_t)g1 * H_ + j0 + s1 * 16 + nn];
    }
  }
  // elementwise-phase assignment: 4 hidden units of one batch row per thread
  const int erow = tid >> 4;
  const int ejl  = (tid & 15) * 4;
  float wout[4], cst[4] = {0.f, 0.f, 0.f, 0.f};
  #pragma unroll
  for (int u = 0; u < 4; ++u) wout[u] = W_out[j0 + ejl + u];

  int target = 0;
  for (int step = 0; step < T_; ++step) {
    const unsigned short* hin  = (step & 1) ? h1 : h0;   // double-buffered h
    unsigned short*       hout = (step & 1) ? h0 : h1;

    // ---- async-stage h tile (16 x 1024 bf16) global -> LDS -----------------
    {
      const unsigned short* hsrc = hin + (size_t)m0 * H_;
      #pragma unroll
      for (int it = 0; it < 8; ++it) {
        int idx = tid + it * 256;                 // 2048 x 16-byte chunks
        int row = idx >> 7, c = idx & 127;
        unsigned ldsa = (unsigned)(uintptr_t)&s_h[row * HPAD + c * 8];
        unsigned long long ga =
            (unsigned long long)(uintptr_t)(hsrc + (size_t)row * H_ + c * 8);
        asm volatile("global_load_async_to_lds_b128 %0, %1, off"
                     :: "v"(ldsa), "v"(ga) : "memory");
      }
#if __has_builtin(__builtin_amdgcn_s_wait_asynccnt)
      __builtin_amdgcn_s_wait_asynccnt(0);
#else
      asm volatile("s_wait_asynccnt 0x0" ::: "memory");
#endif
    }
    __syncthreads();

    // ---- gates = x_gates + h @ W_hh.T  via bf16 WMMA, f32 accumulate -------
    v8f acc0 = xgr0, acc1 = xgr1;
    const unsigned short* hA = s_h + (size_t)nn * HPAD + hi * 8;
    #pragma unroll 2
    for (int kt = 0; kt < 32; ++kt) {
      __builtin_prefetch(wp0 + (size_t)(kt + 1) * 512, 0, 3);
      v8bf alo = *(const v8bf*)(hA + kt * 32);
      v8bf ahi = *(const v8bf*)(hA + kt * 32 + 16);
      v16bf a = __builtin_shufflevector(alo, ahi, 0, 1, 2, 3, 4, 5, 6, 7,
                                                  8, 9, 10, 11, 12, 13, 14, 15);
      v16bf b0 = *(const v16bf*)(wp0 + (size_t)kt * 512);
      v16bf b1 = *(const v16bf*)(wp1 + (size_t)kt * 512);
      acc0 = __builtin_amdgcn_wmma_f32_16x16x32_bf16(false, a, false, b0,
                                                     (short)0, acc0, false, false);
      acc1 = __builtin_amdgcn_wmma_f32_16x16x32_bf16(false, a, false, b1,
                                                     (short)0, acc1, false, false);
    }
    #pragma unroll
    for (int r = 0; r < 8; ++r) {
      int rr = r + hi * 8;
      s_gt[rr][t0 * 16 + nn] = acc0[r];
      s_gt[rr][t1 * 16 + nn] = acc1[r];
    }
    __syncthreads();

    // ---- elementwise LSTM cell update (c in registers across all steps) ----
    float pdot = 0.f;
    #pragma unroll
    for (int u = 0; u < 4; ++u) {
      int jl  = ejl + u;
      float gi = sigm(s_gt[erow][jl]);
      float gf = sigm(s_gt[erow][64 + jl]);
      float gg = ftanh(s_gt[erow][128 + jl]);
      float go = sigm(s_gt[erow][192 + jl]);
      float c  = gf * cst[u] + gi * gg;
      cst[u]   = c;
      float h  = go * ftanh(c);
      hout[(size_t)(m0 + erow) * H_ + j0 + jl] = f2bf(h);
      pdot += h * wout[u];                 // fold O=1 output head into the step
    }
    s_red[erow][tid & 15] = pdot;
    __threadfence();                        // publish h before barrier arrive
    __syncthreads();
    if (tid < 16) {
      float s = 0.f;
      #pragma unroll
      for (int k2 = 0; k2 < 16; ++k2) s += s_red[tid][k2];
      partial[((size_t)(m0 + tid) * T_ + step) * 16 + hc] = s;
    }

    // ---- per-batch-tile barrier across the 16 hidden-chunk WGs -------------
    target += NHC;
    if (tid == 0) {
      __hip_atomic_fetch_add(&cnt[bt], 1, __ATOMIC_RELEASE, __HIP_MEMORY_SCOPE_AGENT);
      while (__hip_atomic_load(&cnt[bt], __ATOMIC_ACQUIRE, __HIP_MEMORY_SCOPE_AGENT) < target)
        __builtin_amdgcn_s_sleep(2);
    }
    __syncthreads();
    __threadfence();                        // acquire: see peers' h stores
  }
}

// ---------------- out[b,t] = b_out + sum_hc partial[b][t][hc] ---------------
__global__ void k_out(const float* __restrict__ partial, const float* __restrict__ b_out,
                      float* __restrict__ out) {
  int i = blockIdx.x * 256 + threadIdx.x;  // 0..16383
  if (i >= B_ * T_) return;
  const float* p = partial + (size_t)i * 16;
  float s = b_out[0];
  #pragma unroll
  for (int k = 0; k < 16; ++k) s += p[k];
  out[i] = s;
}

extern "C" void kernel_launch(void* const* d_in, const int* in_sizes, int n_in,
                              void* d_out, int out_size, void* d_ws, size_t ws_size,
                              hipStream_t stream) {
  const float* latent = (const float*)d_in[0];
  const float* W_lin  = (const float*)d_in[1];
  const float* b_lin  = (const float*)d_in[2];
  const float* W_ih   = (const float*)d_in[3];
  const float* W_hh   = (const float*)d_in[4];
  const float* b_ih   = (const float*)d_in[5];
  const float* b_hh   = (const float*)d_in[6];
  const float* W_out  = (const float*)d_in[7];
  const float* b_out  = (const float*)d_in[8];
  float* out = (float*)d_out;

  char* ws = (char*)d_ws;
  size_t off = 0;
  float* hidden = (float*)(ws + off);          off += (size_t)B_ * H_ * 4;        // 256 KB
  float* xg     = (float*)(ws + off);          off += (size_t)B_ * 4 * H_ * 4;    // 1 MB
  unsigned short* wpk = (unsigned short*)(ws + off); off += (size_t)4 * H_ * H_ * 2; // 8 MB
  unsigned short* h0  = (unsigned short*)(ws + off); off += (size_t)B_ * H_ * 2;  // 128 KB
  unsigned short* h1  = (unsigned short*)(ws + off); off += (size_t)B_ * H_ * 2;  // 128 KB
  float* partial = (float*)(ws + off);         off += (size_t)B_ * T_ * NHC * 4;  // 1 MB
  int* cnt = (int*)(ws + off);                 off += 64;

  k_init  <<<64, 256, 0, stream>>>(cnt, h0);
  k_hidden<<<(B_ * H_) / 256, 256, 0, stream>>>(latent, W_lin, b_lin, hidden);
  k_xgates<<<(B_ * 4 * H_) / 256, 256, 0, stream>>>(hidden, W_ih, b_ih, b_hh, xg);
  k_pack  <<<(4 * H_ * H_) / 256, 256, 0, stream>>>(W_hh, wpk);
  k_lstm  <<<dim3(NHC, NBT), 256, 0, stream>>>(xg, wpk, W_out, h0, h1, partial, cnt);
  k_out   <<<(B_ * T_) / 256, 256, 0, stream>>>(partial, b_out, out);
}
